// FSHNet_nusc_56075093017285
// MI455X (gfx1250) — compile-verified, matrix operand order checked by reading, not run on hardware
//
#include <hip/hip_runtime.h>
#include <hip/hip_bf16.h>

typedef __attribute__((ext_vector_type(16))) _Float16 v16h;
typedef __attribute__((ext_vector_type(8)))  _Float16 v8h;
typedef __attribute__((ext_vector_type(8)))  float    v8f;
typedef __attribute__((ext_vector_type(4)))  unsigned u32x4;
typedef __attribute__((ext_vector_type(8)))  unsigned u32x8;

#define DIMF 128
#define PP   8
#define HH   8
#define HDD  16
#define MPIL 96000
#define PT   16   // pillars per block

// ---- WMMA fragment loader -------------------------------------------------
// 16x32 f16 A-operand (and B-operand, which mirrors A as W^T) from a row-major
// f16 matrix with row stride 128 halves. Per ISA 7.12.2:
//   lane&15 = row, lane>>4 selects K-half; elements 0..7 = K kb+kq*8..+7,
//   elements 8..15 = K kb+16+kq*8..+7  -> two 16-byte loads per lane.
__device__ __forceinline__ v16h load_frag16(const _Float16* base, int rowBase,
                                            int kb, int lane) {
  int r  = rowBase + (lane & 15);
  int kq = (lane >> 4) * 8;
  union { v16h v; uint4 q[2]; } f;
  f.q[0] = *(const uint4*)(base + r * DIMF + kb + kq);
  f.q[1] = *(const uint4*)(base + r * DIMF + kb + 16 + kq);
  return f.v;
}

// pos_emb fragment: row = z = lane&7 (each 16-row M-tile is two pillars, z=0..7)
__device__ __forceinline__ v16h load_posfrag(const _Float16* pos16, int kb,
                                             int lane) {
  int r  = lane & 7;
  int kq = (lane >> 4) * 8;
  union { v16h v; uint4 q[2]; } f;
  f.q[0] = *(const uint4*)(pos16 + r * DIMF + kb + kq);
  f.q[1] = *(const uint4*)(pos16 + r * DIMF + kb + 16 + kq);
  return f.v;
}

// ---- kernel 0: init workspace --------------------------------------------
// grid: 48000 x 256  (i in [0, 12,288,000) == M*DIM == M*P*DIM/8)
__global__ void fshnet_init(const float* __restrict__ w_in,
                            const float* __restrict__ w_out,
                            const float* __restrict__ pos_emb,
                            _Float16* __restrict__ kv16,
                            unsigned* __restrict__ srcbits,
                            int* __restrict__ occ,
                            _Float16* __restrict__ w16,
                            _Float16* __restrict__ wout16,
                            _Float16* __restrict__ pos16) {
  long i = (long)blockIdx.x * 256 + threadIdx.x;
  v8h z = {};
  *(v8h*)(kv16 + i * 8) = z;               // zero kv (M*P*DIM halves)
  srcbits[i] = 0x007FFFFFu;                // mapped(-inf) for float-max trick
  if (i < MPIL)            occ[i]    = 0;
  if (i < 3 * DIMF * DIMF) w16[i]    = (_Float16)w_in[i];
  if (i < DIMF * DIMF)     wout16[i] = (_Float16)w_out[i];
  if (i < PP * DIMF)       pos16[i]  = (_Float16)pos_emb[i];
}

// ---- kernel 1: scatter features into dense (M,P,DIM) f16 + atomic max src --
// grid: N x 128
__global__ void fshnet_scatter(const float* __restrict__ feat,
                               const int* __restrict__ pid,
                               const int* __restrict__ zid,
                               _Float16* __restrict__ kv16,
                               unsigned* __restrict__ srcbits,
                               int* __restrict__ occ) {
  long i = blockIdx.x;
  int  c = threadIdx.x;
  int  p = pid[i], z = zid[i];
  float f = feat[i * DIMF + c];
  kv16[((long)p * PP + z) * DIMF + c] = (_Float16)f;
  unsigned b = __float_as_uint(f);
  unsigned u = (b & 0x80000000u) ? ~b : (b | 0x80000000u); // order-preserving map
  atomicMax(&srcbits[(long)p * DIMF + c], u);
  if (c == 0) atomicOr(&occ[p], 1 << z);
}

// ---- kernel 2: fused q/k/v projection + attention + out proj + LayerNorm ---
// grid: 6000 x 256 (8 wave32), dynamic LDS = 133184 B
__global__ __launch_bounds__(256) void fshnet_fused(
    const float* __restrict__ b_in, const float* __restrict__ b_out,
    const float* __restrict__ gamma, const float* __restrict__ beta,
    const _Float16* __restrict__ kv16, const unsigned* __restrict__ srcbits,
    const int* __restrict__ occ, const _Float16* __restrict__ w16,
    const _Float16* __restrict__ wout16, const _Float16* __restrict__ pos16,
    float* __restrict__ out) {
  extern __shared__ char smem[];
  _Float16* s_vin   = (_Float16*)(smem);           // 128x128 f16 (kv, TDM dest)
  _Float16* s_k     = (_Float16*)(smem + 32768);   // 128x128 f16 (K proj)
  _Float16* s_v     = (_Float16*)(smem + 65536);   // 128x128 f16 (V proj)
  float*    s_q     = (float*)(smem + 98304);      // 16x128 f32 (scaled Q)
  float*    s_src   = (float*)(smem + 106496);     // 16x128 f32
  _Float16* s_src16 = (_Float16*)(smem + 114688);  // 16x128 f16
  _Float16* s_ctx16 = (_Float16*)(smem + 118784);  // 16x128 f16
  float*    s_y     = (float*)(smem + 122880);     // 16x128 f32
  float*    s_sum   = (float*)(smem + 131072);     // 256 f32
  float*    s_sq    = (float*)(smem + 132096);     // 256 f32
  int*      s_occ   = (int*)(smem + 133120);       // 16 ints

  const int tid  = threadIdx.x;
  const int lane = tid & 31;
  const int wave = tid >> 5;
  const int pb   = blockIdx.x * PT;           // pillar base
  const long rowg = (long)pb * PP;            // global kv row base

  // ---- stage 0a: TDM — async-copy the contiguous 32KB kv tile into s_vin.
  // 1-D tile: data_size=8B, tile_dim0 = tensor_dim0 = 4096 elements.
  if (wave == 0) {
    unsigned long long ga = (unsigned long long)(uintptr_t)(kv16 + rowg * DIMF);
    u32x4 g0;
    g0.x = 1u;                                   // count=1, user descriptor
    g0.y = (unsigned)(uintptr_t)s_vin;           // lds_addr (byte offset)
    g0.z = (unsigned)ga;                         // global_addr[31:0]
    g0.w = (unsigned)((ga >> 32) & 0x1FFFFFFu) | (2u << 30);  // addr[56:32]|type=2
    u32x8 g1;
    g1[0] = 0x00030000u;       // wg_mask=0, data_size=3 (8B), no flags
    g1[1] = 0x10000000u;       // atomic_barrier=0 | tensor_dim0.lo16(4096)<<16
    g1[2] = 0x00010000u;       // tensor_dim0.hi16=0 | tensor_dim1.lo16(1)<<16
    g1[3] = 0x10000000u;       // tensor_dim1.hi16=0 | tile_dim0(4096)<<16
    g1[4] = 0u;                // tile_dim1=0 (unused), tile_dim2=0 (unused)
    g1[5] = 4096u;             // tensor_dim0_stride[31:0]
    g1[6] = 0u;                // stride hi | tensor_dim1_stride lo
    g1[7] = 0u;
    u32x4 g2;                  // dims 2/3: length 1, unused tile dims
    g2.x = 1u; g2.y = 1u; g2.z = 0u; g2.w = 0u;
    u32x4 g3;                  // dim4 length 1, tile_dim4 unused
    g3.x = 0u; g3.y = 0x00010000u; g3.z = 0u; g3.w = 0u;
    asm volatile("tensor_load_to_lds %0, %1, %2, %3"
                 :
                 : "s"(g0), "s"(g1), "s"(g2), "s"(g3)
                 : "memory");
  }

  // ---- stage 0b (overlaps TDM): stage src rows + occ into LDS
  {
    int rr = tid >> 4, c0 = (tid & 15) * 8;
#pragma unroll
    for (int i = 0; i < 8; ++i) {
      unsigned u = srcbits[(long)(pb + rr) * DIMF + c0 + i];
      unsigned bb = (u & 0x80000000u) ? (u & 0x7fffffffu) : ~u;  // unmap
      float f = __uint_as_float(bb);
      s_src[rr * DIMF + c0 + i]   = f;
      s_src16[rr * DIMF + c0 + i] = (_Float16)f;
    }
    if (tid < PT) s_occ[tid] = occ[pb + tid];
  }
  if (wave == 0) {
#if __has_builtin(__builtin_amdgcn_s_wait_tensorcnt)
    __builtin_amdgcn_s_wait_tensorcnt(0);
#else
    asm volatile("s_wait_tensorcnt 0x0" ::: "memory");
#endif
  }
  __syncthreads();

  // ---- stage 1: Q projection (wave = N-tile); q = (src @ wq^T + bq) * 0.25
  {
    v8f acc = {};
#pragma unroll
    for (int ks = 0; ks < 4; ++ks) {
      v16h a = load_frag16(s_src16, 0, ks * 32, lane);
      v16h b = load_frag16(w16, wave * 16, ks * 32, lane);   // wq rows = B cols
      acc = __builtin_amdgcn_wmma_f32_16x16x32_f16(false, a, false, b,
                                                   (short)0, acc, false, false);
    }
    int col = wave * 16 + (lane & 15);
    int m0  = (lane >> 4) * 8;
    float bias = b_in[col];
#pragma unroll
    for (int i = 0; i < 8; ++i)
      s_q[(m0 + i) * DIMF + col] = (acc[i] + bias) * 0.25f;  // 1/sqrt(HD)
  }

  // ---- stage 2: K/V projections (wave = 16-row M-tile, 2 pillars)
  //      K A-operand = kv + pos_emb, built in registers (v_pk_add_f16)
  {
    int rb = wave * 16;
    v16h ak[4], av[4];
#pragma unroll
    for (int ks = 0; ks < 4; ++ks) {
      av[ks] = load_frag16(s_vin, rb, ks * 32, lane);
      ak[ks] = av[ks] + load_posfrag(pos16, ks * 32, lane);
    }
    const _Float16* wk16 = w16 + DIMF * DIMF;
    const _Float16* wv16 = w16 + 2 * DIMF * DIMF;
    for (int nt = 0; nt < 8; ++nt) {
      v8f acck = {}, accv = {};
#pragma unroll
      for (int ks = 0; ks < 4; ++ks) {
        v16h bk = load_frag16(wk16, nt * 16, ks * 32, lane);
        v16h bv = load_frag16(wv16, nt * 16, ks * 32, lane);
        acck = __builtin_amdgcn_wmma_f32_16x16x32_f16(false, ak[ks], false, bk,
                                                      (short)0, acck, false, false);
        accv = __builtin_amdgcn_wmma_f32_16x16x32_f16(false, av[ks], false, bv,
                                                      (short)0, accv, false, false);
      }
      int col = nt * 16 + (lane & 15);
      int m0  = (lane >> 4) * 8;
      float bkb = b_in[DIMF + col];
      float bvb = b_in[2 * DIMF + col];
#pragma unroll
      for (int i = 0; i < 8; ++i) {
        int row = rb + m0 + i;
        s_k[row * DIMF + col] = (_Float16)(acck[i] + bkb);
        s_v[row * DIMF + col] = (_Float16)(accv[i] + bvb);
      }
    }
  }
  __syncthreads();

  // ---- stage 3: per-(pillar, head) masked softmax attention (P=8 slots)
  if (tid < PT * HH) {
    int p = tid >> 3, h = tid & 7;
    const float* qp = s_q + p * DIMF + h * HDD;
    float qv[HDD];
#pragma unroll
    for (int d = 0; d < HDD; ++d) qv[d] = qp[d];
    int om = s_occ[p];
    float sc[PP];
    float mx = -3.0e38f;
#pragma unroll
    for (int z = 0; z < PP; ++z) {
      float s = -3.0e38f;
      if ((om >> z) & 1) {
        const _Float16* kp = s_k + (p * PP + z) * DIMF + h * HDD;
        s = 0.f;
#pragma unroll
        for (int d = 0; d < HDD; ++d) s += qv[d] * (float)kp[d];
      }
      sc[z] = s;
      mx = fmaxf(mx, s);
    }
    float den = 0.f;
#pragma unroll
    for (int z = 0; z < PP; ++z) { float e = __expf(sc[z] - mx); sc[z] = e; den += e; }
    float inv = 1.f / den;
    float cx[HDD];
#pragma unroll
    for (int d = 0; d < HDD; ++d) cx[d] = 0.f;
#pragma unroll
    for (int z = 0; z < PP; ++z) {
      float w = sc[z] * inv;
      const _Float16* vp = s_v + (p * PP + z) * DIMF + h * HDD;
#pragma unroll
      for (int d = 0; d < HDD; ++d) cx[d] += w * (float)vp[d];
    }
    _Float16* cp = s_ctx16 + p * DIMF + h * HDD;
#pragma unroll
    for (int d = 0; d < HDD; ++d) cp[d] = (_Float16)cx[d];
  }
  __syncthreads();

  // ---- stage 4: out projection + bias + residual
  {
    v8f acc = {};
#pragma unroll
    for (int ks = 0; ks < 4; ++ks) {
      v16h a = load_frag16(s_ctx16, 0, ks * 32, lane);
      v16h b = load_frag16(wout16, wave * 16, ks * 32, lane);
      acc = __builtin_amdgcn_wmma_f32_16x16x32_f16(false, a, false, b,
                                                   (short)0, acc, false, false);
    }
    int col = wave * 16 + (lane & 15);
    int m0  = (lane >> 4) * 8;
    float bias = b_out[col];
#pragma unroll
    for (int i = 0; i < 8; ++i) {
      int m = m0 + i;
      s_y[m * DIMF + col] = acc[i] + bias + s_src[m * DIMF + col];
    }
  }
  __syncthreads();

  // ---- stage 5: LayerNorm + store (16 threads per row)
  {
    int r = tid >> 4, c0 = (tid & 15) * 8;
    float yv[8], ps = 0.f, pss = 0.f;
#pragma unroll
    for (int i = 0; i < 8; ++i) {
      float y = s_y[r * DIMF + c0 + i];
      yv[i] = y; ps += y; pss += y * y;
    }
    s_sum[tid] = ps; s_sq[tid] = pss;
    __syncthreads();
    float tot = 0.f, tot2 = 0.f;
#pragma unroll
    for (int k = 0; k < 16; ++k) { tot += s_sum[r * 16 + k]; tot2 += s_sq[r * 16 + k]; }
    float mean = tot * (1.f / 128.f);
    float var  = tot2 * (1.f / 128.f) - mean * mean;
    float rstd = rsqrtf(var + 1e-5f);
    float* op = out + (long)(pb + r) * DIMF;
#pragma unroll
    for (int i = 0; i < 8; ++i) {
      int c = c0 + i;
      op[c] = (yv[i] - mean) * rstd * gamma[c] + beta[c];
    }
  }
}

extern "C" void kernel_launch(void* const* d_in, const int* in_sizes, int n_in,
                              void* d_out, int out_size, void* d_ws, size_t ws_size,
                              hipStream_t stream) {
  const float* features  = (const float*)d_in[0];
  const float* w_in      = (const float*)d_in[1];
  const float* b_in      = (const float*)d_in[2];
  const float* w_out     = (const float*)d_in[3];
  const float* b_out     = (const float*)d_in[4];
  const float* pos_emb   = (const float*)d_in[5];
  const float* ln_gamma  = (const float*)d_in[6];
  const float* ln_beta   = (const float*)d_in[7];
  const int*   pillar_id = (const int*)d_in[8];
  const int*   z_id      = (const int*)d_in[9];
  const int N = in_sizes[0] / DIMF;

  // workspace carve (~246.3 MB total)
  char* ws = (char*)d_ws;
  _Float16* kv16 = (_Float16*)ws;                        // M*P*DIM f16
  size_t off = (size_t)MPIL * PP * DIMF * 2;             // 196,608,000
  unsigned* srcbits = (unsigned*)(ws + off);
  off += (size_t)MPIL * DIMF * 4;                        // +49,152,000
  int* occ = (int*)(ws + off);
  off += (size_t)MPIL * 4;                               // +384,000
  off = (off + 15) & ~(size_t)15;
  _Float16* w16 = (_Float16*)(ws + off);   off += 3 * DIMF * DIMF * 2;
  _Float16* wout16 = (_Float16*)(ws + off); off += DIMF * DIMF * 2;
  _Float16* pos16 = (_Float16*)(ws + off); off += PP * DIMF * 2;

  fshnet_init<<<(MPIL * DIMF) / 256, 256, 0, stream>>>(
      w_in, w_out, pos_emb, kv16, srcbits, occ, w16, wout16, pos16);
  fshnet_scatter<<<N, DIMF, 0, stream>>>(features, pillar_id, z_id, kv16,
                                         srcbits, occ);
  fshnet_fused<<<MPIL / PT, 256, 133184, stream>>>(
      b_in, b_out, ln_gamma, ln_beta, kv16, srcbits, occ, w16, wout16, pos16,
      (float*)d_out);
}
